// my_8bit_linear_14748917694589
// MI455X (gfx1250) — compile-verified, matrix operand order checked by reading, not run on hardware
//
#include <hip/hip_runtime.h>

// ---- POD vector types (safe in unions, lower to b64/b128 memory ops) ----
typedef __attribute__((ext_vector_type(16))) _Float16     v16h;
typedef __attribute__((ext_vector_type(8)))  float        v8f;
typedef __attribute__((ext_vector_type(4)))  float        f32x4;
typedef __attribute__((ext_vector_type(4)))  unsigned int u32x4;
typedef __attribute__((ext_vector_type(2)))  unsigned int u32x2;

#define KDIM   1024
#define NDIM   1024
#define TILE_M 128
#define TILE_N 128
#define TILE_K 32
#define LDSK   40                 // padded LDS row stride in halves (80B = 20 banks)
#define NKSTEPS (KDIM / TILE_K)   // 32

union HalfFrag { v16h v; u32x4 q[2]; };

// LDS byte-offset of a __shared__ object (generic -> AS(3) -> 32-bit offset)
__device__ __forceinline__ unsigned lds_offset_of(const void* p) {
  return (unsigned)(unsigned long long)(__attribute__((address_space(3))) const char*)p;
}

// ---------------- CB fp32 -> f16 pre-pass (int8 codes are exact in f16) ---------
__global__ __launch_bounds__(256) void cb_to_f16_kernel(const float* __restrict__ src,
                                                        _Float16* __restrict__ dst,
                                                        int n4) {
  int i = blockIdx.x * 256 + threadIdx.x;
  if (i < n4) {
    f32x4 f = ((const f32x4*)src)[i];
    union { _Float16 h[4]; u32x2 u; } p;
    p.h[0] = (_Float16)f.x; p.h[1] = (_Float16)f.y;
    p.h[2] = (_Float16)f.z; p.h[3] = (_Float16)f.w;
    ((u32x2*)dst)[i] = p.u;
  }
}

// ---------------- main GEMM: 128x128 block tile, f16 WMMA, fp32 epilogue --------
__global__ __launch_bounds__(256) void my_8bit_linear_gemm_kernel(
    const float*    __restrict__ x,     // [M, K] fp32
    const _Float16* __restrict__ cbh,   // [N, K] f16 (pre-converted codes)
    const float*    __restrict__ scb,   // [N]
    const float*    __restrict__ bias,  // [N]
    float*          __restrict__ out)   // [M, N]
{
  __shared__ __attribute__((aligned(16))) _Float16 As[2][TILE_M * LDSK];
  __shared__ __attribute__((aligned(16))) _Float16 Bs[2][TILE_N * LDSK];

  const int tid  = threadIdx.x;
  const int lane = tid & 31;
  const int wave = tid >> 5;
  const int wm   = wave & 3;            // 4 waves along M -> 32 rows each
  const int wn   = wave >> 2;           // 2 waves along N -> 64 cols each
  const int mr   = lane & 15;           // row/col within a 16x16 tile
  const int kh   = (lane >> 4) * 8;     // K sub-offset per ISA A/B f16 layout

  const int blockM = blockIdx.y * TILE_M;
  const int blockN = blockIdx.x * TILE_N;

  // cooperative load indexing: 128 rows x 4 groups of 8 elems = 512 items, 2/thread
  const int ar = tid >> 2;              // 0..63 (+64 for second item)
  const int ag = (tid & 3) * 8;         // K group offset within tile

  const unsigned long long cb_base = (unsigned long long)cbh;

  f32x4 xr[2][2];                       // staged x fp32 (2 items x 8 floats)

  v8f acc[2][4];
#pragma unroll
  for (int i = 0; i < 2; ++i)
#pragma unroll
    for (int j = 0; j < 4; ++j)
      acc[i][j] = (v8f){0.f, 0.f, 0.f, 0.f, 0.f, 0.f, 0.f, 0.f};

  // ---- CDNA5 async byte-copy of the f16 B tile straight into LDS (ASYNCcnt) ----
  auto issue_async_b = [&](int k0, int buf) {
#pragma unroll
    for (int j = 0; j < 2; ++j) {
      unsigned lds_off = lds_offset_of(&Bs[buf][(ar + j * 64) * LDSK + ag]);
      unsigned goff    = (unsigned)(((blockN + ar + j * 64) * KDIM + k0 + ag) *
                                    (int)sizeof(_Float16));
      asm volatile("global_load_async_to_lds_b128 %0, %1, %2"
                   :
                   : "v"(lds_off), "v"(goff), "s"(cb_base)
                   : "memory");
    }
  };

  // ---- global load of one x K-step into registers ----
  auto load_x = [&](int k0) {
#pragma unroll
    for (int j = 0; j < 2; ++j) {
      const float* pa = x + (size_t)(blockM + ar + j * 64) * KDIM + k0 + ag;
      xr[j][0] = *(const f32x4*)pa;
      xr[j][1] = *(const f32x4*)(pa + 4);
    }
  };

  // ---- convert + stage x registers into LDS buffer ----
  auto stage_a = [&](int buf) {
#pragma unroll
    for (int j = 0; j < 2; ++j) {
      union { _Float16 h[8]; u32x4 q; } p;
      p.h[0] = (_Float16)xr[j][0].x; p.h[1] = (_Float16)xr[j][0].y;
      p.h[2] = (_Float16)xr[j][0].z; p.h[3] = (_Float16)xr[j][0].w;
      p.h[4] = (_Float16)xr[j][1].x; p.h[5] = (_Float16)xr[j][1].y;
      p.h[6] = (_Float16)xr[j][1].z; p.h[7] = (_Float16)xr[j][1].w;
      *(u32x4*)&As[buf][(ar + j * 64) * LDSK + ag] = p.q;
    }
  };

  // prologue: fill buffer 0
  issue_async_b(0, 0);
  load_x(0);
  stage_a(0);
  asm volatile("s_wait_asynccnt 0x0" ::: "memory");
  __syncthreads();

#pragma unroll 1
  for (int kk = 0; kk < NKSTEPS; ++kk) {
    const int buf = kk & 1;

    // next tile's loads overlap this iteration's WMMA block:
    //  - B tile streams asynchronously into LDS (no VGPRs)
    //  - x tile lands in registers, converted+staged after the math
    if (kk + 1 < NKSTEPS) {
      issue_async_b((kk + 1) * TILE_K, buf ^ 1);
      load_x((kk + 1) * TILE_K);
    }

    // fragment fetch: two ds_load_b128 per 16x32 fragment (ISA f16 layout)
    HalfFrag a[2], b[4];
#pragma unroll
    for (int sm = 0; sm < 2; ++sm) {
      const int base = (wm * 32 + sm * 16 + mr) * LDSK + kh;
      a[sm].q[0] = *(const u32x4*)&As[buf][base];
      a[sm].q[1] = *(const u32x4*)&As[buf][base + 16];
    }
#pragma unroll
    for (int sn = 0; sn < 4; ++sn) {
      const int base = (wn * 64 + sn * 16 + mr) * LDSK + kh;
      b[sn].q[0] = *(const u32x4*)&Bs[buf][base];
      b[sn].q[1] = *(const u32x4*)&Bs[buf][base + 16];
    }

#pragma unroll
    for (int sm = 0; sm < 2; ++sm)
#pragma unroll
      for (int sn = 0; sn < 4; ++sn)
        acc[sm][sn] = __builtin_amdgcn_wmma_f32_16x16x32_f16(
            /*neg_a=*/false, a[sm].v, /*neg_b=*/false, b[sn].v,
            /*c_mod=*/(short)0, acc[sm][sn],
            /*reuse_a=*/false, /*reuse_b=*/false);

    // stage x into the buffer read next iteration (its readers finished last
    // iteration, so no pre-store barrier is needed with double buffering)
    if (kk + 1 < NKSTEPS) stage_a(buf ^ 1);

    asm volatile("s_wait_asynccnt 0x0" ::: "memory");
    __syncthreads();   // one barrier per K-step
  }

  // ---- fp32 dequant epilogue: out = acc * (SCB/127) + bias ----
  const float inv127 = 1.0f / 127.0f;
#pragma unroll
  for (int sn = 0; sn < 4; ++sn) {
    const int col   = blockN + wn * 64 + sn * 16 + mr;
    const float scl = scb[col] * inv127;
    const float bv  = bias[col];
#pragma unroll
    for (int sm = 0; sm < 2; ++sm) {
      const int row0 = blockM + wm * 32 + sm * 16 + ((lane >> 4) << 3);
#pragma unroll
      for (int r = 0; r < 8; ++r)
        out[(size_t)(row0 + r) * NDIM + col] = acc[sm][sn][r] * scl + bv;
    }
  }
}

extern "C" void kernel_launch(void* const* d_in, const int* in_sizes, int n_in,
                              void* d_out, int out_size, void* d_ws, size_t ws_size,
                              hipStream_t stream) {
  const float* x    = (const float*)d_in[0];   // [B,S,K] fp32
  const float* CB   = (const float*)d_in[1];   // [N,K]   fp32 (int8 codes)
  const float* SCB  = (const float*)d_in[2];   // [N]
  const float* bias = (const float*)d_in[3];   // [N]
  float* out = (float*)d_out;

  const int M = in_sizes[0] / KDIM;            // B*S = 65536
  _Float16* CBh = (_Float16*)d_ws;             // 2 MB scratch

  const int n4 = (NDIM * KDIM) / 4;
  cb_to_f16_kernel<<<n4 / 256, 256, 0, stream>>>(CB, CBh, n4);

  dim3 grid(NDIM / TILE_N, M / TILE_M);        // (8, 512)
  my_8bit_linear_gemm_kernel<<<grid, 256, 0, stream>>>(x, CBh, SCB, bias, out);
}
